// PoseODERNN_73229192397175
// MI455X (gfx1250) — compile-verified, matrix operand order checked by reading, not run on hardware
//
#include <hip/hip_runtime.h>
#include <hip/hip_bf16.h>
#include <stdint.h>

// ---------------- problem constants ----------------
#define BDIM 256   // batch
#define SEQ  32    // sequence length
#define VFD  512
#define IFD  256
#define FD   768   // feature dim (VF+IF)
#define HD   1024  // ode hidden
#define LNUM 2     // rnn layers
#define KSUB 8     // euler substeps

// ---------------- GEMM tiling ----------------
#define BM 64
#define BN 128
#define BK 32
#define NTHR 128  // 4 waves (wave32), 2x2 wave grid, 32x64 per wave
#define LDSS 40   // bf16 elements per LDS row (32 + 8 pad -> 16B-aligned chunks)

typedef __attribute__((ext_vector_type(16))) __bf16    v16bf;
typedef __attribute__((ext_vector_type(8)))  float     v8f;
typedef __attribute__((ext_vector_type(4)))  float     f32x4;
typedef __attribute__((ext_vector_type(4)))  uint32_t  u32x4;

union FragBF { v16bf v; u32x4 q[2]; };

__device__ __forceinline__ unsigned short f32_to_bf16_rne(float f) {
  union { float f; uint32_t u; } c; c.f = f;
  uint32_t u = c.u;
  uint32_t r = u + 0x7FFFu + ((u >> 16) & 1u);
  return (unsigned short)(r >> 16);
}

__device__ __forceinline__ float sigmoidf_fast(float x) {
  return 1.0f / (1.0f + __expf(-x));
}

// CDNA5 async global->LDS DMA (ASYNCcnt-tracked), ISA ch.10 / 15.18.3.
// lds_off: LDS byte offset (generic shared-aperture pointer truncated to 32b,
// valid per ISA 10.2: LDS aperture keeps offset in addr[31:0]).
__device__ __forceinline__ void async_load_b128(uint32_t lds_off, const void* gaddr) {
  asm volatile("global_load_async_to_lds_b128 %0, %1, off"
               :: "v"(lds_off), "v"((uint64_t)(uintptr_t)gaddr)
               : "memory");
}
__device__ __forceinline__ void wait_async0() {
  asm volatile("s_wait_asynccnt 0x0" ::: "memory");
}

// ======================================================================
// Generic WMMA GEMM:  C[M,N] = epilogue( A[M,K] * W[N,K]^T + bias[N] )
// A: fp32 (row stride lda), converted to bf16 on the fly into LDS.
// W: bf16, row-major [N, K] (ldw = K), staged via async-to-LDS DMA.
// modes: 0 = bias only, 1 = tanh, 2 = euler (C = Y + (dt/K)*(acc+bias)),
//        3 = leaky_relu(0.1)
// ======================================================================
__global__ __launch_bounds__(NTHR)
void gemm_bf16_wmma(const float* __restrict__ A, int lda,
                    const unsigned short* __restrict__ W, int ldw,
                    const float* __restrict__ bias,
                    float* __restrict__ C, int ldc,
                    int M, int N, int Kd, int mode,
                    const float* __restrict__ Y,
                    const float* __restrict__ ts,
                    int step, float invK)
{
  __shared__ unsigned short sA[2][BM * LDSS];
  __shared__ unsigned short sB[2][BN * LDSS];

  const int tid   = threadIdx.x;
  const int lane  = tid & 31;
  const int wm    = (tid >> 5) & 1;      // wave M tile (2) -> 32 rows
  const int wn    = (tid >> 6) & 1;      // wave N tile (2) -> 64 cols
  const int lrow  = lane & 15;
  const int khalf = (lane >> 4) << 3;    // 0 or 8 (K-chunk base per ISA A/B layout)
  const int tM    = blockIdx.y * BM;
  const int tN    = blockIdx.x * BN;
  const int nk    = Kd / BK;

  v8f acc[2][4] = {};

  auto stage = [&](int kt, int buf) {
    // ---- A tile: 64x32 fp32 -> bf16, 2x(8 elems)/thread through VGPRs ----
    #pragma unroll
    for (int i = 0; i < 2; ++i) {
      int seg  = tid + i * NTHR;         // 0..255
      int arow = seg >> 2;               // 0..63
      int akg  = (seg & 3) << 3;         // 0,8,16,24
      const float* asrc = A + (size_t)(tM + arow) * lda + (size_t)kt * BK + akg;
      f32x4 f0 = *(const f32x4*)asrc;
      f32x4 f1 = *(const f32x4*)(asrc + 4);
      if (kt + 1 < nk) __builtin_prefetch(asrc + BK, 0, 1);  // global_prefetch_b8
      u32x4 pa;
      pa.x = (uint32_t)f32_to_bf16_rne(f0.x) | ((uint32_t)f32_to_bf16_rne(f0.y) << 16);
      pa.y = (uint32_t)f32_to_bf16_rne(f0.z) | ((uint32_t)f32_to_bf16_rne(f0.w) << 16);
      pa.z = (uint32_t)f32_to_bf16_rne(f1.x) | ((uint32_t)f32_to_bf16_rne(f1.y) << 16);
      pa.w = (uint32_t)f32_to_bf16_rne(f1.z) | ((uint32_t)f32_to_bf16_rne(f1.w) << 16);
      *(u32x4*)(&sA[buf][arow * LDSS + akg]) = pa;
    }
    // ---- B tile: 128x32 bf16, async DMA global->LDS, 4x 16B/thread ----
    #pragma unroll
    for (int i = 0; i < 4; ++i) {
      int seg  = tid + i * NTHR;         // 0..511
      int brow = seg >> 2;               // 0..127
      int bkg  = (seg & 3) << 3;
      const unsigned short* wsrc = W + (size_t)(tN + brow) * ldw + (size_t)kt * BK + bkg;
      uint32_t loff = (uint32_t)(uintptr_t)&sB[buf][brow * LDSS + bkg];
      async_load_b128(loff, wsrc);
    }
  };

  auto compute = [&](int buf) {
    FragBF af[2], bf[4];
    #pragma unroll
    for (int mi = 0; mi < 2; ++mi) {
      const unsigned short* p = &sA[buf][(wm * 32 + mi * 16 + lrow) * LDSS + khalf];
      af[mi].q[0] = *(const u32x4*)p;          // K [c, c+8)
      af[mi].q[1] = *(const u32x4*)(p + 16);   // K [c+16, c+24)
    }
    #pragma unroll
    for (int ni = 0; ni < 4; ++ni) {
      const unsigned short* p = &sB[buf][(wn * 64 + ni * 16 + lrow) * LDSS + khalf];
      bf[ni].q[0] = *(const u32x4*)p;
      bf[ni].q[1] = *(const u32x4*)(p + 16);
    }
    #pragma unroll
    for (int mi = 0; mi < 2; ++mi)
      #pragma unroll
      for (int ni = 0; ni < 4; ++ni)
        acc[mi][ni] = __builtin_amdgcn_wmma_f32_16x16x32_bf16(
            false, af[mi].v, false, bf[ni].v, (short)0, acc[mi][ni], false, false);
  };

  stage(0, 0);
  wait_async0();
  __syncthreads();
  for (int kt = 0; kt < nk; ++kt) {
    int cur = kt & 1;
    if (kt + 1 < nk) stage(kt + 1, cur ^ 1);
    compute(cur);
    wait_async0();         // own async DMAs into next buffer complete
    __syncthreads();       // publish LDS to all waves / protect buffer reuse
  }

  // ---- epilogue: 8 rows per accumulator per lane (ISA C/D layout) ----
  #pragma unroll
  for (int mi = 0; mi < 2; ++mi) {
    #pragma unroll
    for (int ni = 0; ni < 4; ++ni) {
      int col  = tN + wn * 64 + ni * 16 + lrow;
      float bv = bias[col];
      #pragma unroll
      for (int vr = 0; vr < 8; ++vr) {
        int row = tM + wm * 32 + mi * 16 + ((lane >> 4) << 3) + vr;
        float v = acc[mi][ni][vr] + bv;
        float o;
        if (mode == 1) {
          o = tanhf(v);
        } else if (mode == 2) {
          int b = row & (BDIM - 1);
          float dt = (step < SEQ - 1) ? (ts[b * SEQ + step + 1] - ts[b * SEQ + step]) : 0.0f;
          o = Y[(size_t)row * ldc + col] + dt * invK * v;
        } else if (mode == 3) {
          o = (v > 0.0f) ? v : 0.1f * v;
        } else {
          o = v;
        }
        C[(size_t)row * ldc + col] = o;
      }
    }
  }
}

// ---------------- support kernels ----------------
__global__ __launch_bounds__(256)
void cvt_f32_bf16(const float* __restrict__ in, unsigned short* __restrict__ out, int n) {
  int i = blockIdx.x * 256 + threadIdx.x;
  if (i < n) out[i] = f32_to_bf16_rne(in[i]);
}

__global__ __launch_bounds__(256)
void concat_kernel(const float* __restrict__ fv, const float* __restrict__ fi,
                   float* __restrict__ fused, int total) {
  int i = blockIdx.x * 256 + threadIdx.x;
  if (i < total) {
    int f  = i % FD;
    int bs = i / FD;
    fused[i] = (f < VFD) ? fv[(size_t)bs * VFD + f] : fi[(size_t)bs * IFD + (f - VFD)];
  }
}

__global__ __launch_bounds__(256)
void gru_combine_kernel(const float* __restrict__ gx, const float* __restrict__ gh,
                        float* __restrict__ h, float* __restrict__ outs, int t) {
  int i = blockIdx.x * 256 + threadIdx.x;
  if (i < BDIM * FD) {
    int b = i / FD, f = i % FD;
    const float* gxr = gx + (size_t)b * 3 * FD;
    const float* ghr = gh + (size_t)b * 3 * FD;
    float r  = sigmoidf_fast(gxr[f]        + ghr[f]);
    float z  = sigmoidf_fast(gxr[FD + f]   + ghr[FD + f]);
    float n  = tanhf(gxr[2 * FD + f] + r * ghr[2 * FD + f]);
    float hp = h[i];
    float hn = (1.0f - z) * n + z * hp;
    h[i] = hn;
    if (outs) outs[((size_t)b * SEQ + t) * FD + f] = hn;
  }
}

__global__ __launch_bounds__(256)
void pose_head_kernel(const float* __restrict__ hid, const float* __restrict__ W2,
                      const float* __restrict__ b2, float* __restrict__ pose) {
  int i = blockIdx.x * 256 + threadIdx.x;
  if (i < BDIM * SEQ * 6) {
    int m = i / 6, j = i % 6;
    const float* hr = hid + (size_t)m * 128;
    const float* wr = W2  + (size_t)j * 128;
    float s = b2[j];
    #pragma unroll 8
    for (int k = 0; k < 128; ++k) s += hr[k] * wr[k];
    pose[i] = s;
  }
}

// ---------------- host side ----------------
static inline void gemm(hipStream_t st, const float* A, int lda, const unsigned short* W,
                        const float* bias, float* C, int ldc, int M, int N, int Kd, int mode,
                        const float* Y = nullptr, const float* ts = nullptr, int step = 0) {
  dim3 g(N / BN, M / BM), b(NTHR);
  gemm_bf16_wmma<<<g, b, 0, st>>>(A, lda, W, Kd, bias, C, ldc, M, N, Kd, mode,
                                  Y, ts, step, 1.0f / (float)KSUB);
}

extern "C" void kernel_launch(void* const* d_in, const int* in_sizes, int n_in,
                              void* d_out, int out_size, void* d_ws, size_t ws_size,
                              hipStream_t stream) {
  (void)in_sizes; (void)n_in; (void)out_size; (void)ws_size;

  const float* fv     = (const float*)d_in[0];
  const float* fi     = (const float*)d_in[1];
  const float* ts     = (const float*)d_in[2];
  const float* ode_W0 = (const float*)d_in[3];
  const float* ode_b0 = (const float*)d_in[4];
  const float* ode_W1 = (const float*)d_in[5];
  const float* ode_b1 = (const float*)d_in[6];
  const float* ode_W2 = (const float*)d_in[7];
  const float* ode_b2 = (const float*)d_in[8];
  const float* gru_Wx = (const float*)d_in[9];
  const float* gru_Wh = (const float*)d_in[10];
  const float* gru_bx = (const float*)d_in[11];
  const float* gru_bh = (const float*)d_in[12];
  const float* reg_W1 = (const float*)d_in[13];
  const float* reg_b1 = (const float*)d_in[14];
  const float* reg_W2 = (const float*)d_in[15];
  const float* reg_b2 = (const float*)d_in[16];

  char* ws = (char*)d_ws;
  size_t off = 0;
  auto carve = [&](size_t bytes) -> char* {
    off = (off + 255) & ~(size_t)255;
    char* p = ws + off;
    off += bytes;
    return p;
  };

  // bf16 weights (L2-resident working set, converted once per launch)
  unsigned short* wb0 = (unsigned short*)carve((size_t)HD * FD * 2);
  unsigned short* wb1 = (unsigned short*)carve((size_t)HD * HD * 2);
  unsigned short* wb2 = (unsigned short*)carve((size_t)FD * HD * 2);
  unsigned short* wbx = (unsigned short*)carve((size_t)LNUM * 3 * FD * FD * 2);
  unsigned short* wbh = (unsigned short*)carve((size_t)LNUM * 3 * FD * FD * 2);
  unsigned short* wbr = (unsigned short*)carve((size_t)128 * FD * 2);
  // fp32 activations
  float* y     = (float*)carve((size_t)LNUM * BDIM * FD * 4);   // [L,B,F] hidden state
  float* g1    = (float*)carve((size_t)BDIM * 3 * FD * 4);      // gates / ode h1 (aliased)
  float* g2    = (float*)carve((size_t)BDIM * 3 * FD * 4);      // gates / ode h2 (aliased)
  float* fused = (float*)carve((size_t)BDIM * SEQ * FD * 4);    // inputs; reused as outs
  float* hid   = (float*)carve((size_t)BDIM * SEQ * 128 * 4);   // regressor hidden

  // 1) weight conversion fp32 -> bf16
  {
    int n;
    n = HD * FD;            cvt_f32_bf16<<<(n + 255) / 256, 256, 0, stream>>>(ode_W0, wb0, n);
    n = HD * HD;            cvt_f32_bf16<<<(n + 255) / 256, 256, 0, stream>>>(ode_W1, wb1, n);
    n = FD * HD;            cvt_f32_bf16<<<(n + 255) / 256, 256, 0, stream>>>(ode_W2, wb2, n);
    n = LNUM * 3 * FD * FD; cvt_f32_bf16<<<(n + 255) / 256, 256, 0, stream>>>(gru_Wx, wbx, n);
    n = LNUM * 3 * FD * FD; cvt_f32_bf16<<<(n + 255) / 256, 256, 0, stream>>>(gru_Wh, wbh, n);
    n = 128 * FD;           cvt_f32_bf16<<<(n + 255) / 256, 256, 0, stream>>>(reg_W1, wbr, n);
  }

  // 2) fuse inputs, zero hidden state
  int tot = BDIM * SEQ * FD;
  concat_kernel<<<(tot + 255) / 256, 256, 0, stream>>>(fv, fi, fused, tot);
  hipMemsetAsync(y, 0, (size_t)LNUM * BDIM * FD * 4, stream);

  // 3) sequential scan
  for (int s = 0; s < SEQ; ++s) {
    // K Euler substeps of the ODE MLP over the whole [L*B, F] state
    for (int k = 0; k < KSUB; ++k) {
      gemm(stream, y,  FD, wb0, ode_b0, g1, HD, LNUM * BDIM, HD, FD, 1);             // tanh
      gemm(stream, g1, HD, wb1, ode_b1, g2, HD, LNUM * BDIM, HD, HD, 1);             // tanh
      gemm(stream, g2, HD, wb2, ode_b2, y,  FD, LNUM * BDIM, FD, HD, 2, y, ts, s);   // y += (dt/K)*f
    }
    // GRU stack
    for (int l = 0; l < LNUM; ++l) {
      const float* xin = (l == 0) ? (fused + (size_t)s * FD) : y;   // layer1 input = updated h0
      int xlda         = (l == 0) ? SEQ * FD : FD;
      gemm(stream, xin, xlda, wbx + (size_t)l * 3 * FD * FD, gru_bx + (size_t)l * 3 * FD,
           g1, 3 * FD, BDIM, 3 * FD, FD, 0);
      gemm(stream, y + (size_t)l * BDIM * FD, FD, wbh + (size_t)l * 3 * FD * FD,
           gru_bh + (size_t)l * 3 * FD, g2, 3 * FD, BDIM, 3 * FD, FD, 0);
      gru_combine_kernel<<<(BDIM * FD + 255) / 256, 256, 0, stream>>>(
          g1, g2, y + (size_t)l * BDIM * FD, (l == LNUM - 1) ? fused : nullptr, s);
    }
  }

  // 4) regressor: [B*S,768] -> leaky_relu -> [B*S,128] -> [B*S,6]
  gemm(stream, fused, FD, wbr, reg_b1, hid, 128, BDIM * SEQ, 128, FD, 3);
  pose_head_kernel<<<(BDIM * SEQ * 6 + 255) / 256, 256, 0, stream>>>(
      hid, reg_W2, reg_b2, (float*)d_out);

  // 5) h_last -> second output
  hipMemcpyAsync((float*)d_out + (size_t)BDIM * SEQ * 6, y,
                 (size_t)LNUM * BDIM * FD * 4, hipMemcpyDeviceToDevice, stream);
}
